// MoELayer_49606872268988
// MI455X (gfx1250) — compile-verified, matrix operand order checked by reading, not run on hardware
//
#include <hip/hip_runtime.h>

typedef __attribute__((ext_vector_type(16))) __bf16 v16bf;
typedef __attribute__((ext_vector_type(8)))  float  v8f;

#define Dq 1024
#define Fq 4096
#define Eq 8

// ---------------------------------------------------------------- WMMA helper
__device__ __forceinline__ v8f wmma_bf16(v16bf a, v16bf b, v8f c) {
#if defined(__gfx1250__)
  return __builtin_amdgcn_wmma_f32_16x16x32_bf16(false, a, false, b, (short)0, c,
                                                 false, false);
#else
  return c;
#endif
}

// Async copy of 32 contiguous bytes global -> LDS (2 x B128), ASYNCcnt-tracked.
// LDS address = low 32 bits of the flat shared-aperture pointer (ISA 10.2).
__device__ __forceinline__ void async_copy32(const void* gsrc, void* ldst) {
#if defined(__gfx1250__)
  unsigned l = (unsigned)(unsigned long long)(size_t)ldst;
  asm volatile("global_load_async_to_lds_b128 %0, %1, off"
               :: "v"(l), "v"(gsrc) : "memory");
  asm volatile("global_load_async_to_lds_b128 %0, %1, off offset:16"
               :: "v"(l), "v"(gsrc) : "memory");
#else
  const __bf16* s = (const __bf16*)gsrc;
  __bf16* d = (__bf16*)ldst;
  for (int i = 0; i < 16; ++i) d[i] = s[i];
#endif
}

__device__ __forceinline__ void wait_async0() {
#if defined(__gfx1250__)
  asm volatile("s_wait_asynccnt 0" ::: "memory");
#endif
}

// Load a 16x32 bf16 fragment from an LDS tile with leading dim 32.
// Lane L (<16): row = row0+L, k = {0..7, 16..23}; lane L (>=16): row = row0+L-16,
// k = {8..15, 24..31}  (CDNA5 ISA 16-bit A-matrix layout; B assumed symmetric).
__device__ __forceinline__ v16bf load_frag(const __bf16* tile, int row0, int lane) {
  int r  = row0 + (lane & 15);
  int kg = (lane >> 4) * 8;
  const __bf16* p = tile + r * 32 + kg;
  v16bf v;
#pragma unroll
  for (int i = 0; i < 8; ++i) v[i] = p[i];
#pragma unroll
  for (int i = 0; i < 8; ++i) v[8 + i] = p[16 + i];
  return v;
}

// ---------------------------------------------------------------- init
__global__ void init_stats_kernel(float* stats) {
  if (threadIdx.x < 16) stats[threadIdx.x] = 0.f;
}

// ---------------------------------------------------------------- router
__global__ void __launch_bounds__(128) router_kernel(
    const float* __restrict__ xf, const float* __restrict__ gate_w,
    int* __restrict__ idx, float* __restrict__ rw, float* __restrict__ stats,
    int T) {
  __shared__ float gs[Eq * Dq];
  int tid = threadIdx.x;
  for (int i = tid; i < Eq * Dq; i += blockDim.x) gs[i] = gate_w[i];
  __syncthreads();

  int wid = tid >> 5, lane = tid & 31;
  int t = blockIdx.x * 4 + wid;
  if (t >= T) return;

  const float* xr = xf + (size_t)t * Dq;
  float acc[Eq];
#pragma unroll
  for (int e = 0; e < Eq; ++e) acc[e] = 0.f;
  for (int d = lane; d < Dq; d += 32) {
    float xv = xr[d];
#pragma unroll
    for (int e = 0; e < Eq; ++e) acc[e] += xv * gs[e * Dq + d];
  }
#pragma unroll
  for (int off = 16; off >= 1; off >>= 1)
#pragma unroll
    for (int e = 0; e < Eq; ++e) acc[e] += __shfl_xor(acc[e], off, 32);

  if (lane == 0) {
    int i0 = 0;
#pragma unroll
    for (int e = 1; e < Eq; ++e)
      if (acc[e] > acc[i0]) i0 = e;
    int i1 = (i0 == 0) ? 1 : 0;
#pragma unroll
    for (int e = 0; e < Eq; ++e)
      if (e != i0 && acc[e] > acc[i1]) i1 = e;
    float l0 = acc[i0], l1 = acc[i1];
    float e1 = __expf(l1 - l0);
    float inv = 1.f / (1.f + e1);
    rw[2 * t]     = inv;
    rw[2 * t + 1] = e1 * inv;
    idx[2 * t]     = i0;
    idx[2 * t + 1] = i1;
    atomicAdd(&stats[i0], 1.f);
    atomicAdd(&stats[i1], 1.f);
    float mx = acc[0];
#pragma unroll
    for (int e = 1; e < Eq; ++e) mx = fmaxf(mx, acc[e]);
    float p[Eq], s = 0.f;
#pragma unroll
    for (int e = 0; e < Eq; ++e) { p[e] = __expf(acc[e] - mx); s += p[e]; }
    float invs = 1.f / s;
#pragma unroll
    for (int e = 0; e < Eq; ++e) atomicAdd(&stats[8 + e], p[e] * invs);
  }
}

// ---------------------------------------------------------------- rank / capacity
__global__ void __launch_bounds__(1024) rank_kernel(
    const int* __restrict__ idx, float* __restrict__ rw, int* __restrict__ srow,
    int total, int Ccap) {
  __shared__ int e_sh[1024];
  __shared__ int base[Eq];
  __shared__ int cnt[Eq];
  int tid = threadIdx.x;
  if (tid < Eq) base[tid] = 0;
  __syncthreads();
  for (int start = 0; start < total; start += 1024) {
    int s = start + tid;
    int e = (s < total) ? idx[s] : -1;
    e_sh[tid] = e;
    if (tid < Eq) cnt[tid] = 0;
    __syncthreads();
    if (s < total) {
      int r = 0;
      for (int j = 0; j < tid; ++j) r += (e_sh[j] == e);
      int rank = base[e] + r;
      bool valid = rank < Ccap;
      srow[s] = valid ? e * Ccap + rank : Eq * Ccap;
      if (!valid) rw[s] = 0.f;
    }
    if (tid < Eq) {
      int c = 0;
      for (int j = 0; j < 1024; ++j) c += (e_sh[j] == tid);
      cnt[tid] = c;
    }
    __syncthreads();
    if (tid < Eq) base[tid] += cnt[tid];
    __syncthreads();
  }
}

// ---------------------------------------------------------------- dispatch
__global__ void __launch_bounds__(256) dispatch_kernel(
    const float* __restrict__ xf, const int* __restrict__ srow,
    __bf16* __restrict__ eb, int Ccap) {
  int s = blockIdx.x;
  int row = srow[s];
  if (row >= Eq * Ccap) return;
  int t = s >> 1;
  const float* src = xf + (size_t)t * Dq;
  __bf16* dst = eb + (size_t)row * Dq;
  for (int i = threadIdx.x; i < Dq; i += blockDim.x) dst[i] = (__bf16)src[i];
}

// ---------------------------------------------------------------- FFN GEMM 1
// H[c,f] = silu(X.W1^T) * (X.W3^T) per expert.  Double-buffered pipeline:
// X tiles move global->LDS via async B128 copies; W1/W3 fp32 tiles stage
// through registers (cvt to bf16) into the alternate LDS buffer while WMMAs
// consume the current one.  One barrier per k-step.
__global__ void __launch_bounds__(256) ffn1_kernel(
    const __bf16* __restrict__ eb, const float* __restrict__ w1,
    const float* __restrict__ w3, __bf16* __restrict__ H, int Ccap) {
  int e = blockIdx.z;
  int c0 = blockIdx.y * 128;
  int f0 = blockIdx.x * 128;
  __shared__ __bf16 Xs[2][128 * 32];
  __shared__ __bf16 W1s[2][128 * 32];
  __shared__ __bf16 W3s[2][128 * 32];

  int tid = threadIdx.x;
  int lane = tid & 31;
  int wid = tid >> 5;
  int wm = wid & 3;   // 4 row groups of 32
  int wn = wid >> 2;  // 2 col groups of 64

  int ldr = tid >> 1;        // 0..127 tile row
  int ldo = (tid & 1) * 16;  // 0 / 16 within the 32-wide k chunk
  int grc = c0 + ldr;
  if (grc > Ccap - 1) grc = Ccap - 1;  // clamp; padded rows discarded in epilogue

  const __bf16* Xrow  = eb + ((size_t)e * Ccap + grc) * Dq + ldo;
  const float*  W1row = w1 + ((size_t)e * Fq + f0 + ldr) * Dq + ldo;
  const float*  W3row = w3 + ((size_t)e * Fq + f0 + ldr) * Dq + ldo;

  const v8f vzero = {0.f, 0.f, 0.f, 0.f, 0.f, 0.f, 0.f, 0.f};
  v8f acc1[2][4], acc3[2][4];
#pragma unroll
  for (int i = 0; i < 2; ++i)
#pragma unroll
    for (int j = 0; j < 4; ++j) { acc1[i][j] = vzero; acc3[i][j] = vzero; }

  float r1[16], r3[16];
  int lofs = ldr * 32 + ldo;

  // ---- issue tile kk into buffer buf (async X + W regs)
  auto issue = [&](int kk, int buf) {
    async_copy32(Xrow + kk, &Xs[buf][lofs]);
#pragma unroll
    for (int i = 0; i < 16; ++i) r1[i] = W1row[kk + i];
#pragma unroll
    for (int i = 0; i < 16; ++i) r3[i] = W3row[kk + i];
  };
  // ---- convert staged weights into buffer buf
  auto commit = [&](int buf) {
    __bf16* d1 = &W1s[buf][lofs];
    __bf16* d3 = &W3s[buf][lofs];
#pragma unroll
    for (int i = 0; i < 16; ++i) { d1[i] = (__bf16)r1[i]; d3[i] = (__bf16)r3[i]; }
  };

  issue(0, 0);
  commit(0);
  wait_async0();
  __syncthreads();

  for (int kk = 0; kk < Dq; kk += 32) {
    int cur = (kk >> 5) & 1;
    int nxt = cur ^ 1;
    bool more = (kk + 32) < Dq;
    if (more) issue(kk + 32, nxt);

    v16bf a0 = load_frag(&Xs[cur][0], wm * 32 + 0,  lane);
    v16bf a1 = load_frag(&Xs[cur][0], wm * 32 + 16, lane);
#pragma unroll
    for (int j = 0; j < 4; ++j) {
      v16bf b1 = load_frag(&W1s[cur][0], wn * 64 + j * 16, lane);
      v16bf b3 = load_frag(&W3s[cur][0], wn * 64 + j * 16, lane);
      acc1[0][j] = wmma_bf16(a0, b1, acc1[0][j]);
      acc1[1][j] = wmma_bf16(a1, b1, acc1[1][j]);
      acc3[0][j] = wmma_bf16(a0, b3, acc3[0][j]);
      acc3[1][j] = wmma_bf16(a1, b3, acc3[1][j]);
    }

    if (more) { commit(nxt); wait_async0(); }
    __syncthreads();
  }

  int rbase = (lane >> 4) * 8;
  int col_l = lane & 15;
#pragma unroll
  for (int i = 0; i < 2; ++i)
#pragma unroll
    for (int j = 0; j < 4; ++j) {
      int rowt = c0 + wm * 32 + i * 16 + rbase;
      int colt = f0 + wn * 64 + j * 16 + col_l;
#pragma unroll
      for (int r = 0; r < 8; ++r) {
        int row = rowt + r;
        if (row < Ccap) {
          float v1 = acc1[i][j][r], v3 = acc3[i][j][r];
          float h = v1 / (1.f + __expf(-v1)) * v3;
          H[((size_t)e * Ccap + row) * Fq + colt] = (__bf16)h;
        }
      }
    }
}

// ---------------------------------------------------------------- FFN GEMM 2
// eo[c,d] = H . W2^T per expert (K = F).  Same pipelined structure.
__global__ void __launch_bounds__(256) ffn2_kernel(
    const __bf16* __restrict__ H, const float* __restrict__ w2,
    float* __restrict__ eo, int Ccap) {
  int e = blockIdx.z;
  int c0 = blockIdx.y * 128;
  int d0 = blockIdx.x * 128;
  __shared__ __bf16 Hs[2][128 * 32];
  __shared__ __bf16 W2s[2][128 * 32];

  int tid = threadIdx.x;
  int lane = tid & 31;
  int wid = tid >> 5;
  int wm = wid & 3;
  int wn = wid >> 2;

  int ldr = tid >> 1;
  int ldo = (tid & 1) * 16;
  int grc = c0 + ldr;
  if (grc > Ccap - 1) grc = Ccap - 1;

  const __bf16* Hrow  = H + ((size_t)e * Ccap + grc) * Fq + ldo;
  const float*  W2row = w2 + ((size_t)e * Dq + d0 + ldr) * Fq + ldo;

  const v8f vzero = {0.f, 0.f, 0.f, 0.f, 0.f, 0.f, 0.f, 0.f};
  v8f acc[2][4];
#pragma unroll
  for (int i = 0; i < 2; ++i)
#pragma unroll
    for (int j = 0; j < 4; ++j) acc[i][j] = vzero;

  float r2[16];
  int lofs = ldr * 32 + ldo;

  auto issue = [&](int kk, int buf) {
    async_copy32(Hrow + kk, &Hs[buf][lofs]);
#pragma unroll
    for (int i = 0; i < 16; ++i) r2[i] = W2row[kk + i];
  };
  auto commit = [&](int buf) {
    __bf16* d2 = &W2s[buf][lofs];
#pragma unroll
    for (int i = 0; i < 16; ++i) d2[i] = (__bf16)r2[i];
  };

  issue(0, 0);
  commit(0);
  wait_async0();
  __syncthreads();

  for (int kk = 0; kk < Fq; kk += 32) {
    int cur = (kk >> 5) & 1;
    int nxt = cur ^ 1;
    bool more = (kk + 32) < Fq;
    if (more) issue(kk + 32, nxt);

    v16bf a0 = load_frag(&Hs[cur][0], wm * 32 + 0,  lane);
    v16bf a1 = load_frag(&Hs[cur][0], wm * 32 + 16, lane);
#pragma unroll
    for (int j = 0; j < 4; ++j) {
      v16bf b = load_frag(&W2s[cur][0], wn * 64 + j * 16, lane);
      acc[0][j] = wmma_bf16(a0, b, acc[0][j]);
      acc[1][j] = wmma_bf16(a1, b, acc[1][j]);
    }

    if (more) { commit(nxt); wait_async0(); }
    __syncthreads();
  }

  int rbase = (lane >> 4) * 8;
  int col_l = lane & 15;
#pragma unroll
  for (int i = 0; i < 2; ++i)
#pragma unroll
    for (int j = 0; j < 4; ++j) {
      int rowt = c0 + wm * 32 + i * 16 + rbase;
      int colt = d0 + wn * 64 + j * 16 + col_l;
#pragma unroll
      for (int r = 0; r < 8; ++r) {
        int row = rowt + r;
        if (row < Ccap)
          eo[((size_t)e * Ccap + row) * Dq + colt] = acc[i][j][r];
      }
    }
}

// ---------------------------------------------------------------- combine
__global__ void __launch_bounds__(256) combine_kernel(
    const float* __restrict__ eo, const int* __restrict__ srow,
    const float* __restrict__ rw, float* __restrict__ y, int Ccap) {
  int t = blockIdx.x;
  int r0 = srow[2 * t], r1 = srow[2 * t + 1];
  float w0 = rw[2 * t], w1 = rw[2 * t + 1];
  int dump = Eq * Ccap;
  for (int d = threadIdx.x; d < Dq; d += blockDim.x) {
    float v = 0.f;
    if (r0 < dump) v += w0 * eo[(size_t)r0 * Dq + d];
    if (r1 < dump) v += w1 * eo[(size_t)r1 * Dq + d];
    y[(size_t)t * Dq + d] = v;
  }
}

// ---------------------------------------------------------------- aux loss
__global__ void aux_kernel(const float* __restrict__ stats, float* __restrict__ out,
                           int T) {
  if (threadIdx.x == 0 && blockIdx.x == 0) {
    float invT = 1.f / (float)T;
    float s = 0.f;
#pragma unroll
    for (int e = 0; e < Eq; ++e) s += (stats[e] * invT) * (stats[8 + e] * invT);
    out[0] = 0.01f * (float)Eq * s;
  }
}

// ---------------------------------------------------------------- host launch
extern "C" void kernel_launch(void* const* d_in, const int* in_sizes, int n_in,
                              void* d_out, int out_size, void* d_ws, size_t ws_size,
                              hipStream_t stream) {
  (void)n_in; (void)out_size; (void)ws_size;
  const float* x  = (const float*)d_in[0];
  const float* gw = (const float*)d_in[1];
  const float* w1 = (const float*)d_in[2];
  const float* w3 = (const float*)d_in[3];
  const float* w2 = (const float*)d_in[4];
  float* y = (float*)d_out;

  size_t T = (size_t)in_sizes[0] / Dq;
  long long cap = ((long long)T * 5 + 4 * Eq - 1) / (4 * Eq);  // ceil(T*1.25/E)
  if (cap < 1) cap = 1;
  if (cap > (long long)T) cap = (long long)T;
  int C = (int)cap;

  char* ws = (char*)d_ws;
  auto aln = [](size_t v) { return (v + 255) & ~(size_t)255; };
  size_t o = 0;
  float* rw    = (float*)(ws + o); o = aln(o + T * 2 * sizeof(float));
  int*   idx   = (int*)(ws + o);   o = aln(o + T * 2 * sizeof(int));
  int*   srow  = (int*)(ws + o);   o = aln(o + T * 2 * sizeof(int));
  float* stats = (float*)(ws + o); o = aln(o + 16 * sizeof(float));
  __bf16* eb   = (__bf16*)(ws + o); o = aln(o + (size_t)Eq * C * Dq * sizeof(__bf16));
  __bf16* Hb   = (__bf16*)(ws + o); o = aln(o + (size_t)Eq * C * Fq * sizeof(__bf16));
  float*  eo   = (float*)(ws + o);  o = aln(o + (size_t)Eq * C * Dq * sizeof(float));

  init_stats_kernel<<<1, 32, 0, stream>>>(stats);
  router_kernel<<<dim3((unsigned)((T + 3) / 4)), 128, 0, stream>>>(x, gw, idx, rw,
                                                                   stats, (int)T);
  rank_kernel<<<1, 1024, 0, stream>>>(idx, rw, srow, (int)(T * 2), C);
  dispatch_kernel<<<dim3((unsigned)(T * 2)), 256, 0, stream>>>(x, srow, eb, C);

  dim3 g1(Fq / 128, (unsigned)((C + 127) / 128), Eq);
  ffn1_kernel<<<g1, 256, 0, stream>>>(eb, w1, w3, Hb, C);
  dim3 g2(Dq / 128, (unsigned)((C + 127) / 128), Eq);
  ffn2_kernel<<<g2, 256, 0, stream>>>(Hb, w2, eo, C);

  combine_kernel<<<dim3((unsigned)T), 256, 0, stream>>>(eo, srow, rw, y, C);
  aux_kernel<<<1, 32, 0, stream>>>(stats, y + T * (size_t)Dq, (int)T);
}
